// Attention_64106681860425
// MI455X (gfx1250) — matrix operand order repaired and checked by reading, not compiled, not run on hardware
//
#include <hip/hip_runtime.h>
#include <hip/hip_bf16.h>

typedef __attribute__((ext_vector_type(16))) _Float16 v16h;
typedef __attribute__((ext_vector_type(8)))  _Float16 v8h;
typedef __attribute__((ext_vector_type(8)))  float    v8f;

#define HW 4096
#define HH 64
#define WW 64

#if __has_builtin(__builtin_amdgcn_global_load_async_to_lds_b128) && \
    __has_builtin(__builtin_amdgcn_s_wait_asynccnt)
#define USE_ASYNC_LDS 1
typedef int v4i_ll __attribute__((vector_size(16)));
typedef __attribute__((address_space(1))) v4i_ll glb_v4i;
typedef __attribute__((address_space(3))) v4i_ll lds_v4i;
#else
#define USE_ASYNC_LDS 0
#endif

// ---------------------------------------------------------------- elementwise
__global__ void ew_add_kernel(const float* __restrict__ a, const float* __restrict__ b,
                              float* __restrict__ o, int n) {
  int i = blockIdx.x * blockDim.x + threadIdx.x;
  if (i < n) o[i] = a[i] + b[i];
}

// pack OIHW f32 weights -> [dy][cblk][dx][co][ci(32)] f16 (fragment-ready, contiguous per staging)
__global__ void pack_w_kernel(const float* __restrict__ src, _Float16* __restrict__ dst,
                              int Cout, int Cin, int ks) {
  int i = blockIdx.x * blockDim.x + threadIdx.x;
  int KK = ks * ks;
  if (i >= Cout * Cin * KK) return;
  int kx = i % ks; int tmp = i / ks;
  int ky = tmp % ks; tmp /= ks;
  int ci = tmp % Cin; int co = tmp / Cin;
  int cb = ci >> 5, cl = ci & 31;
  int cblocks = Cin >> 5;
  size_t o = ((((size_t)(ky * cblocks + cb) * ks + kx) * Cout + co) << 5) + cl;
  dst[o] = (_Float16)src[i];
}

// ---------------------------------------------------------------- global pool
__global__ void pool_kernel(const float* __restrict__ xx, float* __restrict__ ymean,
                            float* __restrict__ ymax) {
  int bc = blockIdx.x;
  const float* src = xx + (size_t)bc * HW;
  __shared__ float ssum[256], smax[256];
  int t = threadIdx.x;
  float s = 0.f, mx = -1e30f;
  for (int i = t; i < HW; i += 256) { float v = src[i]; s += v; mx = fmaxf(mx, v); }
  ssum[t] = s; smax[t] = mx; __syncthreads();
  for (int k = 128; k > 0; k >>= 1) {
    if (t < k) { ssum[t] += ssum[t + k]; smax[t] = fmaxf(smax[t], smax[t + k]); }
    __syncthreads();
  }
  if (t == 0) { ymean[bc] = ssum[0] * (1.f / 4096.f); ymax[bc] = smax[0]; }
}

// ---------------------------------------------------------------- MCAM (one block per batch)
__global__ void mcam_kernel(const float* __restrict__ ymean, const float* __restrict__ ymax,
                            const float* mc0w, const float* mc0b,
                            const float* mc1w, const float* mc1b,
                            const float* mc2w, const float* mc2b,
                            const float* mc3w, const float* mc3b,
                            const float* mccw, const float* mccb,
                            const float* cse1w, const float* cse1b,
                            const float* cse2w, const float* cse2b,
                            const float* fcw, const float* fcb,
                            const float* fc1w, const float* fc1s, const float* fc1t,
                            float* __restrict__ gout, float* __restrict__ proxy) {
  __shared__ float yfull[2048];
  __shared__ float yc[2048];
  __shared__ float gsh[256];
  __shared__ float hsh[64];
  __shared__ float cf[19];
  __shared__ float y1sh[19];
  __shared__ float cm[256 * 19];
  int b = blockIdx.x, t = threadIdx.x;
  const float* srcm = ymean + b * 256;
  const float* srcx = ymax + b * 256;
  const float* wlist[4] = {mc0w, mc1w, mc2w, mc3w};
  const float* blist[4] = {mc0b, mc1b, mc2b, mc3b};
  const int kszs[4] = {5, 9, 17, 33};
  for (int k = 0; k < 8; ++k) {
    int e = t + (k << 8);
    int j = e >> 8, i = e & 255;
    const float* src = (j < 4) ? srcm : srcx;
    int jj = j & 3;
    int K = kszs[jj], pad = (K - 1) >> 1;
    const float* w = wlist[jj];
    float acc = blist[jj][0];
    for (int tt = 0; tt < K; ++tt) {
      int idx = i + tt - pad;
      if (idx >= 0 && idx < 256) acc += w[tt] * src[idx];
    }
    yfull[e] = acc;
  }
  __syncthreads();
  for (int k = 0; k < 8; ++k) {
    int e = t + (k << 8);
    float acc = mccb[0];
    for (int tt = 0; tt < 3; ++tt) {
      int idx = e + tt - 1;
      if (idx >= 0 && idx < 2048) acc += mccw[tt] * yfull[idx];
    }
    yc[e] = acc;
  }
  __syncthreads();
  {
    float acc = fcb[t];
    const float* wr = fcw + (size_t)t * 2048;
    for (int i = 0; i < 2048; ++i) acc += yc[i] * wr[i];
    float gv = 1.f / (1.f + expf(-acc));
    gsh[t] = gv;
    gout[b * 256 + t] = gv;
  }
  __syncthreads();
  if (t < 64) {
    float acc = cse1b[t];
    const float* wr = cse1w + t * 256;
    for (int c = 0; c < 256; ++c) acc += wr[c] * gsh[c];
    hsh[t] = fmaxf(acc, 0.f);
  }
  __syncthreads();
  if (t < 19) {
    float acc = cse2b[t];
    const float* wr = cse2w + t * 64;
    for (int j = 0; j < 64; ++j) acc += wr[j] * hsh[j];
    y1sh[t] = 1.f / (1.f + expf(-acc));
    float f = 0.f;
    const float* wr2 = fc1w + t * 256;
    for (int c = 0; c < 256; ++c) f += wr2[c] * gsh[c];
    cf[t] = f * fc1s[t] + fc1t[t];
  }
  __syncthreads();
  if (t == 0) {
    float m = -1e30f;
    for (int n = 0; n < 19; ++n) m = fmaxf(m, cf[n]);
    float s = 0.f;
    for (int n = 0; n < 19; ++n) { cf[n] = expf(cf[n] - m); s += cf[n]; }
    for (int n = 0; n < 19; ++n) cf[n] = cf[n] / s + y1sh[n];
  }
  __syncthreads();
  for (int n = 0; n < 19; ++n) cm[t * 19 + n] = gsh[t] * cf[n];
  __syncthreads();
  if (t < 19) {
    float m = -1e30f;
    for (int c = 0; c < 256; ++c) m = fmaxf(m, cm[c * 19 + t]);
    float s = 0.f;
    for (int c = 0; c < 256; ++c) s += expf(cm[c * 19 + t] - m);
    float inv = 1.f / s;
    for (int c = 0; c < 256; ++c) cm[c * 19 + t] = expf(cm[c * 19 + t] - m) * inv;
  }
  __syncthreads();
  {
    float m = -1e30f;
    for (int n = 0; n < 19; ++n) m = fmaxf(m, cm[t * 19 + n]);
    float s = 0.f; float e_[19];
    for (int n = 0; n < 19; ++n) { e_[n] = expf(cm[t * 19 + n] - m); s += e_[n]; }
    float inv = 1.f / s;
    for (int n = 0; n < 19; ++n) proxy[((size_t)b * 256 + t) * 19 + n] = e_[n] * inv;
  }
}

// ---------------------------------------------------------------- MSAM attention, writes f16
__global__ void msam_kernel(const float* __restrict__ xx, _Float16* __restrict__ out) {
  int bc = blockIdx.x;
  const float* src = xx + (size_t)bc * HW;
  __shared__ float xh[64];
  __shared__ float xw[64];
  __shared__ float red[256];
  int t = threadIdx.x;
  if (t < 64) {
    float s = 0.f;
    for (int y = 0; y < 64; ++y) s += src[y * 64 + t];
    xh[t] = s * (1.f / 64.f);
  } else if (t < 128) {
    int y = t - 64;
    float s = 0.f;
    for (int x = 0; x < 64; ++x) s += src[y * 64 + x];
    xw[y] = s * (1.f / 64.f);
  }
  __syncthreads();
  float lmax = -1e30f;
  for (int i = t; i < HW; i += 256) lmax = fmaxf(lmax, xw[i >> 6] * xh[i & 63]);
  red[t] = lmax; __syncthreads();
  for (int k = 128; k > 0; k >>= 1) { if (t < k) red[t] = fmaxf(red[t], red[t + k]); __syncthreads(); }
  float m = red[0]; __syncthreads();
  float lsum = 0.f;
  for (int i = t; i < HW; i += 256) lsum += expf(xw[i >> 6] * xh[i & 63] - m);
  red[t] = lsum; __syncthreads();
  for (int k = 128; k > 0; k >>= 1) { if (t < k) red[t] += red[t + k]; __syncthreads(); }
  float inv = 1.f / red[0];
  for (int i = t; i < HW; i += 256) {
    float a = expf(xw[i >> 6] * xh[i & 63] - m) * inv;
    out[(size_t)bc * HW + i] = (_Float16)(src[i] * (a + 1.f));
  }
}

// ---------------------------------------------------------------- K/V 1x1 convs on the 19-wide proxy
__global__ void kv_kernel(const float* __restrict__ proxy,
                          const float* fkw, const float* fks, const float* fkt,
                          const float* fvw, const float* fvs, const float* fvt,
                          float* __restrict__ kmat, float* __restrict__ vmat) {
  int b = blockIdx.x, t = threadIdx.x;
  __shared__ float pr[256 * 19];
  for (int i = t; i < 256 * 19; i += 256) pr[i] = proxy[(size_t)b * 256 * 19 + i];
  __syncthreads();
  for (int e = t; e < 128 * 19; e += 256) {
    int kc = e / 19, n = e - kc * 19;
    float a0 = 0.f, a1 = 0.f;
    const float* wk = fkw + kc * 256;
    const float* wv = fvw + kc * 256;
    for (int c = 0; c < 256; ++c) { float p = pr[c * 19 + n]; a0 += wk[c] * p; a1 += wv[c] * p; }
    kmat[(size_t)b * 128 * 19 + e] = fminf(fmaxf(a0 * fks[kc] + fkt[kc], 0.f), 6.f);
    vmat[(size_t)b * 128 * 19 + e] = fminf(fmaxf(a1 * fvs[kc] + fvt[kc], 0.f), 6.f);
  }
}

// ---------------------------------------------------------------- qk^T logits + softmax over hw
__global__ void attn_softmax_kernel(const _Float16* __restrict__ q, const float* __restrict__ kmat,
                                    float* __restrict__ mm) {
  int n = blockIdx.x, b = blockIdx.y, t = threadIdx.x;
  __shared__ float kcol[128];
  __shared__ float red[256];
  if (t < 128) kcol[t] = kmat[((size_t)b * 128 + t) * 19 + n];
  __syncthreads();
  const float scale = 0.08838834764831845f;  // 128^-0.5
  float l[16];
  for (int i = 0; i < 16; ++i) {
    int p = t + (i << 8);
    const _Float16* qp = q + (size_t)b * 128 * HW + p;
    float acc = 0.f;
    for (int kc = 0; kc < 128; ++kc) acc += (float)qp[(size_t)kc * HW] * kcol[kc];
    l[i] = acc * scale;
  }
  float mx = l[0];
  for (int i = 1; i < 16; ++i) mx = fmaxf(mx, l[i]);
  red[t] = mx; __syncthreads();
  for (int k = 128; k > 0; k >>= 1) { if (t < k) red[t] = fmaxf(red[t], red[t + k]); __syncthreads(); }
  mx = red[0]; __syncthreads();
  float s = 0.f;
  for (int i = 0; i < 16; ++i) { l[i] = expf(l[i] - mx); s += l[i]; }
  red[t] = s; __syncthreads();
  for (int k = 128; k > 0; k >>= 1) { if (t < k) red[t] += red[t + k]; __syncthreads(); }
  float inv = 1.f / red[0];
  float* dst = mm + ((size_t)b * 19 + n) * HW;
  for (int i = 0; i < 16; ++i) dst[t + (i << 8)] = l[i] * inv;
}

// ---------------------------------------------------------------- ctx = softmax(mm) @ v, writes f16
__global__ void attn_ctx_kernel(const float* __restrict__ mm, const float* __restrict__ vmat,
                                _Float16* __restrict__ ctx) {
  int kc = blockIdx.x, b = blockIdx.y, t = threadIdx.x;
  __shared__ float vrow[19];
  if (t < 19) vrow[t] = vmat[((size_t)b * 128 + kc) * 19 + t];
  __syncthreads();
  for (int p = t; p < HW; p += 256) {
    float acc = 0.f;
    for (int n = 0; n < 19; ++n) acc += mm[((size_t)b * 19 + n) * HW + p] * vrow[n];
    ctx[((size_t)b * 128 + kc) * HW + p] = (_Float16)acc;
  }
}

// ---------------------------------------------------------------- WMMA implicit-GEMM conv
// Block: one image row (64 px) x 128 output channels. 8 waves: m-subtile = wv&3,
// n-group = wv>>2 (64 co each), 4 accumulators per wave. Outer loop (dy, ci-block):
// stage 66x32 halo A-tile (covers all 3 dx) + KS contiguous 128x32 weight blocks
// (via GLOBAL_LOAD_ASYNC_TO_LDS_B128 when available; ASYNCcnt wait before barrier).
// A frag (lane l, g=l/16): j<8 -> K=j+8g, j>=8 -> K=j+8+8g. B frag: same as A.
template <int KS>
__global__ void conv_wmma_kernel(const _Float16* __restrict__ inA, const _Float16* __restrict__ inAdd,
                                 const _Float16* __restrict__ inB2,
                                 const _Float16* __restrict__ wgt,
                                 const float* __restrict__ scale, const float* __restrict__ shift,
                                 _Float16* __restrict__ outH, float* __restrict__ outF,
                                 int Cin, int Cin1, int Cout) {
  const int PAD = KS / 2;
  const int XW = 64 + 2 * PAD;
  __shared__ __align__(16) _Float16 As[66 * 32];       // [xp][ci]
  __shared__ __align__(16) _Float16 Bs[3 * 128 * 32];  // [dx][co][ci]
  int t = threadIdx.x;
  int co0 = blockIdx.x << 7;
  int y = blockIdx.y, b = blockIdx.z;
  int wv = t >> 5, lane = t & 31, g = lane >> 4, l15 = lane & 15;
  int m0 = (wv & 3) << 4;
  int nsg = wv >> 2;
  int cblocks = Cin >> 5;
  int C2 = Cin - Cin1;
  v8f acc[4];
#pragma unroll
  for (int jn = 0; jn < 4; ++jn) acc[jn] = (v8f){0.f, 0.f, 0.f, 0.f, 0.f, 0.f, 0.f, 0.f};

  for (int dy = 0; dy < KS; ++dy) {
    int yy = y + dy - PAD;
    for (int cb = 0; cb < cblocks; ++cb) {
      int c0 = cb << 5;
      // ---- stage B: KS blocks of 128x32 halves (contiguous copies)
#pragma unroll
      for (int dx = 0; dx < KS; ++dx) {
        const _Float16* srcB = wgt + ((((size_t)(dy * cblocks + cb) * KS + dx) * Cout + co0) << 5);
#if USE_ASYNC_LDS
        for (int i = t; i < 512; i += 256) {
          glb_v4i* gp = (glb_v4i*)(uintptr_t)(const void*)(srcB + (size_t)i * 8);
          lds_v4i* lp = (lds_v4i*)(uint32_t)(uintptr_t)(void*)(&Bs[dx * 4096 + i * 8]);
          __builtin_amdgcn_global_load_async_to_lds_b128(gp, lp, 0, 0);
        }
#else
        const uint4* s4 = (const uint4*)srcB;
        uint4* d4 = (uint4*)&Bs[dx * 4096];
        for (int i = t; i < 512; i += 256) d4[i] = s4[i];
#endif
      }
      // ---- stage A halo tile (XW x 32), f16 global -> f16 LDS (pad/transpose/add)
      for (int idx = t; idx < XW * 32; idx += 256) {
        int xp = idx >> 5, ci = idx & 31;
        int x = xp - PAD;
        float v = 0.f;
        if ((unsigned)x < 64u && (unsigned)yy < 64u) {
          int cg = c0 + ci;
          if (cg < Cin1) {
            size_t a = (((size_t)b * Cin1 + cg) << 12) + (yy << 6) + x;
            v = (float)inA[a];
            if (inAdd) v += (float)inAdd[a];
            if (cb + 1 < cblocks && idx == t) __builtin_prefetch(inA + a + (32u << 12), 0, 1);
          } else {
            size_t a = (((size_t)b * C2 + (cg - Cin1)) << 12) + (yy << 6) + x;
            v = (float)inB2[a];
          }
        }
        As[(xp << 5) + ci] = (_Float16)v;
      }
#if USE_ASYNC_LDS
      __builtin_amdgcn_s_wait_asynccnt(0);
#endif
      __syncthreads();
      // ---- 12 (KS=3) / 4 (KS=1) WMMA per staging round per wave
#pragma unroll
      for (int dx = 0; dx < KS; ++dx) {
        const _Float16* ap = &As[(m0 + l15 + dx) << 5];
        v8h alo = *(const v8h*)(ap + 8 * g);
        v8h ahi = *(const v8h*)(ap + 16 + 8 * g);
        v16h af = __builtin_shufflevector(alo, ahi, 0, 1, 2, 3, 4, 5, 6, 7,
                                          8, 9, 10, 11, 12, 13, 14, 15);
#pragma unroll
        for (int jn = 0; jn < 4; ++jn) {
          const _Float16* bp = &Bs[(dx << 12) + (((nsg << 6) + (jn << 4) + l15) << 5) + 8 * g];
          v8h blo = *(const v8h*)bp;
          v8h bhi = *(const v8h*)(bp + 16);
          v16h bf = __builtin_shufflevector(blo, bhi, 0, 1, 2, 3, 4, 5, 6, 7,
                                            8, 9, 10, 11, 12, 13, 14, 15);
          acc[jn] = __builtin_amdgcn_wmma_f32_16x16x32_f16(false, af, false, bf,
                                                           (short)0, acc[jn], false, false);
        }
      }
      __syncthreads();
    }
  }
  // ---- epilogue: BN scale/shift + ReLU6; D layout: VGPR r -> M=m0+r+8g, N=n0+l15
#pragma unroll
  for (int jn = 0; jn < 4; ++jn) {
    int n = co0 + (nsg << 6) + (jn << 4) + l15;
    float sc = scale[n], sh = shift[n];
#pragma unroll
    for (int r = 0; r < 8; ++r) {
      int m = m0 + r + (g << 3);
      float v = acc[jn][r] * sc + sh;
      v = fminf(fmaxf(v, 0.f), 6.f);
      size_t off = (((size_t)b * Cout + n) << 12) + (y << 6) + m;
      if (outH) outH[off] = (_Float16)v;
      else outF[off] = v;
    }
  }
}

// ---------------------------------------------------------------- launch
extern "C" void kernel_launch(void* const* d_in, const int* in_sizes, int n_in,
                              void* d_out, int out_size, void* d_ws, size_t ws_size,
                              hipStream_t stream) {
  const float* x    = (const float*)d_in[0];
  const float* skip = (const float*)d_in[1];
  const float* mc0w = (const float*)d_in[2];  const float* mc0b = (const float*)d_in[3];
  const float* mc1w = (const float*)d_in[4];  const float* mc1b = (const float*)d_in[5];
  const float* mc2w = (const float*)d_in[6];  const float* mc2b = (const float*)d_in[7];
  const float* mc3w = (const float*)d_in[8];  const float* mc3b = (const float*)d_in[9];
  const float* mccw = (const float*)d_in[10]; const float* mccb = (const float*)d_in[11];
  const float* cse1w = (const float*)d_in[12]; const float* cse1b = (const float*)d_in[13];
  const float* cse2w = (const float*)d_in[14]; const float* cse2b = (const float*)d_in[15];
  const float* fcw  = (const float*)d_in[16]; const float* fcb  = (const float*)d_in[17];
  const float* fc1w = (const float*)d_in[18]; const float* fc1s = (const float*)d_in[19];
  const float* fc1t = (const float*)d_in[20];
  const float* msw  = (const float*)d_in[21]; const float* mss  = (const float*)d_in[22];
  const float* mst  = (const float*)d_in[23];
  const float* fqw  = (const float*)d_in[24]; const float* fqs  = (const float*)d_in[25];
  const float* fqt  = (const float*)d_in[26];
  const float* fkw  = (const float*)d_in[27]; const float* fks  = (const float*)d_in[28];
  const float* fkt  = (const float*)d_in[29];
  const float* fvw  = (const float*)d_in[30]; const float* fvs  = (const float*)d_in[31];
  const float* fvt  = (const float*)d_in[32];
  const float* fupw = (const float*)d_in[33]; const float* fups = (const float*)d_in[34];
  const float* fupt = (const float*)d_in[35];
  const float* fusew = (const float*)d_in[36]; const float* fuses = (const float*)d_in[37];
  const float* fuset = (const float*)d_in[38];
  const float* c3w  = (const float*)d_in[39]; const float* c3s  = (const float*)d_in[40];
  const float* c3t  = (const float*)d_in[41];
  const float* c1w  = (const float*)d_in[42]; const float* c1s  = (const float*)d_in[43];
  const float* c1t  = (const float*)d_in[44];

  const size_t SZ_FULL = 8ull * 256 * HW;   // 8,388,608
  const size_t SZ_HALF = 8ull * 128 * HW;   // 4,194,304
  // ---- f32 region
  float* W = (float*)d_ws;
  float* xx    = W;                          // SZ_FULL
  float* mm    = xx + SZ_FULL;               // 8*19*4096
  float* ymean = mm + 8ull * 19 * HW;
  float* ymax  = ymean + 2048;
  float* gbuf  = ymax + 2048;
  float* proxy = gbuf + 2048;                // 8*256*19
  float* kmat  = proxy + 8 * 256 * 19;       // 8*128*19
  float* vmat  = kmat + 8 * 128 * 19;
  float* endf  = vmat + 8 * 128 * 19;
  // ---- f16 region
  _Float16* msam_in = (_Float16*)endf;       // SZ_FULL
  _Float16* sa_h    = msam_in + SZ_FULL;     // SZ_FULL
  _Float16* q_h     = sa_h + SZ_FULL;        // SZ_HALF
  _Float16* ctx_h   = q_h + SZ_HALF;         // SZ_HALF
  _Float16* ctxu_h  = ctx_h + SZ_HALF;       // SZ_FULL
  _Float16* fuse_o  = ctxu_h + SZ_FULL;      // SZ_FULL
  _Float16* sp_h    = fuse_o + SZ_FULL;      // SZ_FULL
  _Float16* ms_p    = sp_h + SZ_FULL;
  _Float16* fq_p    = ms_p + 256 * 256 * 9;
  _Float16* fup_p   = fq_p + 128 * 256 * 9;
  _Float16* fuse_p  = fup_p + 256 * 128;
  _Float16* c3_p    = fuse_p + 256 * 256 * 9;
  _Float16* c1_p    = c3_p + 256 * 256 * 9;

  // weight pre-pack (f32 OIHW -> fragment-ready f16)
  {
    struct { const float* s; _Float16* d; int co, ci, ks; } pk[6] = {
      {msw, ms_p, 256, 256, 3}, {fqw, fq_p, 128, 256, 3}, {fupw, fup_p, 256, 128, 1},
      {fusew, fuse_p, 256, 256, 3}, {c3w, c3_p, 256, 256, 3}, {c1w, c1_p, 256, 512, 1}};
    for (int i = 0; i < 6; ++i) {
      int n = pk[i].co * pk[i].ci * pk[i].ks * pk[i].ks;
      pack_w_kernel<<<(n + 255) / 256, 256, 0, stream>>>(pk[i].s, pk[i].d, pk[i].co, pk[i].ci,
                                                         pk[i].ks);
    }
  }

  int nfull = (int)SZ_FULL;
  ew_add_kernel<<<(nfull + 255) / 256, 256, 0, stream>>>(skip, x, xx, nfull);
  pool_kernel<<<2048, 256, 0, stream>>>(xx, ymean, ymax);
  mcam_kernel<<<8, 256, 0, stream>>>(ymean, ymax, mc0w, mc0b, mc1w, mc1b, mc2w, mc2b,
                                     mc3w, mc3b, mccw, mccb, cse1w, cse1b, cse2w, cse2b,
                                     fcw, fcb, fc1w, fc1s, fc1t, gbuf, proxy);
  msam_kernel<<<2048, 256, 0, stream>>>(xx, msam_in);
  // sa = cbr(ms 3x3)
  conv_wmma_kernel<3><<<dim3(2, 64, 8), 256, 0, stream>>>(msam_in, nullptr, nullptr, ms_p,
                                                          mss, mst, sa_h, nullptr, 256, 256, 256);
  kv_kernel<<<8, 256, 0, stream>>>(proxy, fkw, fks, fkt, fvw, fvs, fvt, kmat, vmat);
  // q = cbr(fq 3x3)
  conv_wmma_kernel<3><<<dim3(1, 64, 8), 256, 0, stream>>>(sa_h, nullptr, nullptr, fq_p,
                                                          fqs, fqt, q_h, nullptr, 256, 256, 128);
  attn_softmax_kernel<<<dim3(19, 8), 256, 0, stream>>>(q_h, kmat, mm);
  attn_ctx_kernel<<<dim3(128, 8), 256, 0, stream>>>(mm, vmat, ctx_h);
  // ctxu = cbr(fup 1x1)
  conv_wmma_kernel<1><<<dim3(2, 64, 8), 256, 0, stream>>>(ctx_h, nullptr, nullptr, fup_p,
                                                          fups, fupt, ctxu_h, nullptr, 128, 128, 256);
  // fuse_out = cbr(fuse 3x3) on (ctxu + sa)
  conv_wmma_kernel<3><<<dim3(2, 64, 8), 256, 0, stream>>>(ctxu_h, sa_h, nullptr, fuse_p,
                                                          fuses, fuset, fuse_o, nullptr, 256, 256, 256);
  // sp = cbr(c3 3x3) on sa
  conv_wmma_kernel<3><<<dim3(2, 64, 8), 256, 0, stream>>>(sa_h, nullptr, nullptr, c3_p,
                                                          c3s, c3t, sp_h, nullptr, 256, 256, 256);
  // out = cbr(c1 1x1) on concat[fuse_out, sp] -> f32
  conv_wmma_kernel<1><<<dim3(2, 64, 8), 256, 0, stream>>>(fuse_o, nullptr, sp_h, c1_p,
                                                          c1s, c1t, nullptr, (float*)d_out, 512, 256, 256);
}
